// MoEPredictor_89524298318522
// MI455X (gfx1250) — matrix operand order repaired and checked by reading, not compiled, not run on hardware
//
#include <hip/hip_runtime.h>
#include <hip/hip_bf16.h>
#include <math.h>

// ---------------------------------------------------------------------------
// MoE predictor for MI455X (gfx1250, wave32, WMMA).  All heavy GEMMs run on
// v_wmma_f32_16x16x32_bf16; top-2 routing is exploited by compacting tokens
// per expert so we do ~450 GFLOP instead of the reference's dense 1.7 TFLOP.
// ---------------------------------------------------------------------------

typedef __bf16 bf16;
typedef __bf16 v16bf __attribute__((ext_vector_type(16)));
typedef __bf16 v4bf  __attribute__((ext_vector_type(4)));
typedef float  v8f   __attribute__((ext_vector_type(8)));
typedef float  v4f   __attribute__((ext_vector_type(4)));

#define N_TOK  8192
#define DMODEL 1024
#define DFF    4096
#define NEXP   8
#define EPSV   1.1920928955078125e-07f   // FLT_EPSILON (torch RMSNorm default)

struct FragBF { v16bf v; };

// A-matrix 16x32 bf16 fragment (ISA 7.12.2): lane half 0 -> K {0..7,16..23},
// lane half 1 -> K {8..15,24..31}.  p points at (row, k + hs*8).
__device__ __forceinline__ void load_frag_a(FragBF& f, const bf16* __restrict__ p) {
    v4f* q = reinterpret_cast<v4f*>(&f.v);
    q[0] = *reinterpret_cast<const v4f*>(p);        // K chunk +0 (8 bf16)
    q[1] = *reinterpret_cast<const v4f*>(p + 16);   // K chunk +16
}
// B-matrix 32x16 bf16 fragment: lane n holds column n (= row n of row-major W),
// K {0..7,16..23} (half 0) / {8..15,24..31} (half 1).  p at (ncol, k + hs*8).
__device__ __forceinline__ void load_frag_b(FragBF& f, const bf16* __restrict__ p) {
    v4f* q = reinterpret_cast<v4f*>(&f.v);
    q[0] = *reinterpret_cast<const v4f*>(p);
    q[1] = *reinterpret_cast<const v4f*>(p + 16);
}
__device__ __forceinline__ v8f wmma_bf16(const FragBF& a, const FragBF& b, v8f c) {
    return __builtin_amdgcn_wmma_f32_16x16x32_bf16(false, a.v, false, b.v,
                                                   (short)0, c, false, false);
}

// ---------------------------------------------------------------------------
// fp32 -> bf16 bulk conversion (RNE), 4 elems / thread / iter
// ---------------------------------------------------------------------------
__global__ __launch_bounds__(256) void f32_to_bf16_kernel(
    const float* __restrict__ src, bf16* __restrict__ dst, long n4) {
    long i  = (long)blockIdx.x * blockDim.x + threadIdx.x;
    long st = (long)gridDim.x * blockDim.x;
    for (; i < n4; i += st) {
        v4f f = reinterpret_cast<const v4f*>(src)[i];
        v4bf o;
        o[0] = (bf16)f[0]; o[1] = (bf16)f[1]; o[2] = (bf16)f[2]; o[3] = (bf16)f[3];
        reinterpret_cast<v4bf*>(dst)[i] = o;
    }
}

__global__ __launch_bounds__(256) void zero_f32_kernel(float* __restrict__ p, long n) {
    long i  = (long)blockIdx.x * blockDim.x + threadIdx.x;
    long st = (long)gridDim.x * blockDim.x;
    for (; i < n; i += st) p[i] = 0.f;
}

// ---------------------------------------------------------------------------
// C(f32, MxN) = A(bf16, MxK row-major) @ W(bf16, NxK row-major)^T
// 256 threads = 8 waves (2x4), block tile 128x128, wave tile 64x32.
// cnt_p != nullptr => row limit read from device (compacted expert rows).
// ---------------------------------------------------------------------------
__global__ __launch_bounds__(256) void gemm_bf16_kernel(
    const bf16* __restrict__ A, int lda,
    const bf16* __restrict__ W, int ldw,
    float* __restrict__ C, int ldc,
    int M, int Kdim, const int* __restrict__ cnt_p)
{
    int Mlim = M;
    if (cnt_p) { int c = *cnt_p; Mlim = c < M ? c : M; }
    if ((int)blockIdx.y * 128 >= Mlim) return;

    const int lane = threadIdx.x & 31;
    const int wave = threadIdx.x >> 5;
    const int lr = lane & 15;
    const int hs = (lane >> 4) & 1;
    const int mw = wave & 1;
    const int nw = wave >> 1;
    const int m0 = blockIdx.y * 128 + mw * 64;
    const int n0 = blockIdx.x * 128 + nw * 32;

    const v8f vz = {};
    v8f acc[4][2];
#pragma unroll
    for (int t = 0; t < 4; ++t)
#pragma unroll
        for (int u = 0; u < 2; ++u) acc[t][u] = vz;

    for (int k = 0; k < Kdim; k += 32) {
        FragBF a[4], b[2];
#pragma unroll
        for (int t = 0; t < 4; ++t)
            load_frag_a(a[t], A + (size_t)(m0 + t * 16 + lr) * lda + k + hs * 8);
#pragma unroll
        for (int u = 0; u < 2; ++u)
            load_frag_b(b[u], W + (size_t)(n0 + u * 16 + lr) * ldw + k + hs * 8);
#pragma unroll
        for (int t = 0; t < 4; ++t)
#pragma unroll
            for (int u = 0; u < 2; ++u)
                acc[t][u] = wmma_bf16(a[t], b[u], acc[t][u]);
    }

#pragma unroll
    for (int t = 0; t < 4; ++t)
#pragma unroll
        for (int u = 0; u < 2; ++u) {
            const int col  = n0 + u * 16 + lr;
            const int rowb = m0 + t * 16 + hs * 8;
#pragma unroll
            for (int v = 0; v < 8; ++v) {
                const int r = rowb + v;
                if (r < Mlim) C[(size_t)r * ldc + col] = acc[t][u][v];
            }
        }
}

// ---------------------------------------------------------------------------
// Fused expert gate/up GEMM with token gather:
//   G[r,:] = silu(h[tok[r]] @ Wg_e^T) * (h[tok[r]] @ Wu_e^T)   (bf16 out)
// Shares the A fragment between both WMMA chains (same 128x128 tiling).
// ---------------------------------------------------------------------------
__global__ __launch_bounds__(256) void moe_gateup_kernel(
    const bf16* __restrict__ H,
    const bf16* __restrict__ Wg, const bf16* __restrict__ Wu,
    bf16* __restrict__ G,
    const int* __restrict__ tok, const int* __restrict__ cnt_p)
{
    const int cnt = *cnt_p;
    if ((int)blockIdx.y * 128 >= cnt) return;

    const int lane = threadIdx.x & 31;
    const int wave = threadIdx.x >> 5;
    const int lr = lane & 15;
    const int hs = (lane >> 4) & 1;
    const int mw = wave & 1;
    const int nw = wave >> 1;
    const int m0 = blockIdx.y * 128 + mw * 64;
    const int n0 = blockIdx.x * 128 + nw * 32;

    int rowsrc[4];
#pragma unroll
    for (int t = 0; t < 4; ++t) {
        const int r = m0 + t * 16 + lr;
        rowsrc[t] = (r < cnt) ? tok[r] : 0;
    }

    const v8f vz = {};
    v8f accg[4][2], accu[4][2];
#pragma unroll
    for (int t = 0; t < 4; ++t)
#pragma unroll
        for (int u = 0; u < 2; ++u) { accg[t][u] = vz; accu[t][u] = vz; }

    for (int k = 0; k < DMODEL; k += 32) {
        FragBF a[4], bg[2], bu[2];
#pragma unroll
        for (int t = 0; t < 4; ++t)
            load_frag_a(a[t], H + (size_t)rowsrc[t] * DMODEL + k + hs * 8);
#pragma unroll
        for (int u = 0; u < 2; ++u) {
            const size_t boff = (size_t)(n0 + u * 16 + lr) * DMODEL + k + hs * 8;
            load_frag_b(bg[u], Wg + boff);
            load_frag_b(bu[u], Wu + boff);
        }
#pragma unroll
        for (int t = 0; t < 4; ++t)
#pragma unroll
            for (int u = 0; u < 2; ++u) {
                accg[t][u] = wmma_bf16(a[t], bg[u], accg[t][u]);
                accu[t][u] = wmma_bf16(a[t], bu[u], accu[t][u]);
            }
    }

#pragma unroll
    for (int t = 0; t < 4; ++t)
#pragma unroll
        for (int u = 0; u < 2; ++u) {
            const int col  = n0 + u * 16 + lr;
            const int rowb = m0 + t * 16 + hs * 8;
#pragma unroll
            for (int v = 0; v < 8; ++v) {
                const int r = rowb + v;
                if (r < cnt) {
                    const float g  = accg[t][u][v];
                    const float up = accu[t][u][v];
                    const float sg = g / (1.f + __expf(-g));   // silu
                    G[(size_t)r * DFF + col] = (bf16)(sg * up);
                }
            }
        }
}

// ---------------------------------------------------------------------------
// Fused input RMSNorm + router: writes bf16 h, computes 8 logits via
// deterministic tree reduction, top-2 on (logits + bias), softmax gates over
// original logits, compacts token ids + gates per expert.
// ---------------------------------------------------------------------------
__global__ __launch_bounds__(256) void rmsnorm_router_kernel(
    const float* __restrict__ Hpre, const float* __restrict__ norm_w,
    const float* __restrict__ Wr, const float* __restrict__ bias,
    bf16* __restrict__ Hbf,
    int* __restrict__ counts, int* __restrict__ tok, float* __restrict__ gate)
{
    const int n = blockIdx.x;
    const int t = threadIdx.x;
    const float* x = Hpre + (size_t)n * DMODEL;
    __shared__ float red[256];
    __shared__ float lbuf[NEXP * 256];

    float v[4];
    float ss = 0.f;
#pragma unroll
    for (int j = 0; j < 4; ++j) { v[j] = x[t + j * 256]; ss += v[j] * v[j]; }
    red[t] = ss; __syncthreads();
    for (int s2 = 128; s2 > 0; s2 >>= 1) { if (t < s2) red[t] += red[t + s2]; __syncthreads(); }
    const float rms = rsqrtf(red[0] * (1.f / DMODEL) + EPSV);

    float part[NEXP];
#pragma unroll
    for (int e = 0; e < NEXP; ++e) part[e] = 0.f;
#pragma unroll
    for (int j = 0; j < 4; ++j) {
        const int i = t + j * 256;
        const float h = v[j] * rms * norm_w[i];
        Hbf[(size_t)n * DMODEL + i] = (bf16)h;
#pragma unroll
        for (int e = 0; e < NEXP; ++e) part[e] += h * Wr[e * DMODEL + i];
    }
#pragma unroll
    for (int e = 0; e < NEXP; ++e) lbuf[e * 256 + t] = part[e];
    __syncthreads();
    for (int s2 = 128; s2 > 0; s2 >>= 1) {
        if (t < s2)
#pragma unroll
            for (int e = 0; e < NEXP; ++e) lbuf[e * 256 + t] += lbuf[e * 256 + t + s2];
        __syncthreads();
    }
    if (t == 0) {
        float lg[NEXP], bi[NEXP];
#pragma unroll
        for (int e = 0; e < NEXP; ++e) { lg[e] = lbuf[e * 256]; bi[e] = lg[e] + bias[e]; }
        int i0 = 0;
#pragma unroll
        for (int e = 1; e < NEXP; ++e) if (bi[e] > bi[i0]) i0 = e;      // first-max tie rule
        int i1 = (i0 == 0) ? 1 : 0;
#pragma unroll
        for (int e = 0; e < NEXP; ++e)
            if (e != i0 && e != i1 && bi[e] > bi[i1]) i1 = e;
        const float m  = fmaxf(lg[i0], lg[i1]);
        const float e0 = __expf(lg[i0] - m), e1 = __expf(lg[i1] - m);
        const float inv = 1.f / (e0 + e1);
        const int s0 = atomicAdd(&counts[i0], 1);
        tok[(size_t)i0 * N_TOK + s0]  = n;
        gate[(size_t)i0 * N_TOK + s0] = e0 * inv;
        const int s1 = atomicAdd(&counts[i1], 1);
        tok[(size_t)i1 * N_TOK + s1]  = n;
        gate[(size_t)i1 * N_TOK + s1] = e1 * inv;
    }
}

// ---------------------------------------------------------------------------
// Per-expert epilogue: out[tok[r]] += gate[r] * rmsnorm(Y[r], enorm_w_e)
// (experts run sequentially => no write conflicts, bit-deterministic)
// ---------------------------------------------------------------------------
__global__ __launch_bounds__(256) void expert_epilogue_kernel(
    const float* __restrict__ Y, const int* __restrict__ tok,
    const float* __restrict__ gate, const int* __restrict__ cnt_p,
    const float* __restrict__ enw, float* __restrict__ outacc)
{
    const int r = blockIdx.x;
    if (r >= *cnt_p) return;
    const int t = threadIdx.x;
    const float* y = Y + (size_t)r * DMODEL;
    __shared__ float red[256];
    float v[4];
    float ss = 0.f;
#pragma unroll
    for (int j = 0; j < 4; ++j) { v[j] = y[t + j * 256]; ss += v[j] * v[j]; }
    red[t] = ss; __syncthreads();
    for (int s2 = 128; s2 > 0; s2 >>= 1) { if (t < s2) red[t] += red[t + s2]; __syncthreads(); }
    const float rms = rsqrtf(red[0] * (1.f / DMODEL) + EPSV);
    const int   token = tok[r];
    const float g     = gate[r];
    float* o = outacc + (size_t)token * DMODEL;
#pragma unroll
    for (int j = 0; j < 4; ++j) { const int i = t + j * 256; o[i] += g * v[j] * rms * enw[i]; }
}

// ---------------------------------------------------------------------------
// Final RMSNorm -> bf16 (input to the W_out GEMM)
// ---------------------------------------------------------------------------
__global__ __launch_bounds__(256) void rmsnorm_to_bf16_kernel(
    const float* __restrict__ X, const float* __restrict__ w, bf16* __restrict__ out)
{
    const int n = blockIdx.x;
    const int t = threadIdx.x;
    const float* x = X + (size_t)n * DMODEL;
    __shared__ float red[256];
    float v[4];
    float ss = 0.f;
#pragma unroll
    for (int j = 0; j < 4; ++j) { v[j] = x[t + j * 256]; ss += v[j] * v[j]; }
    red[t] = ss; __syncthreads();
    for (int s2 = 128; s2 > 0; s2 >>= 1) { if (t < s2) red[t] += red[t + s2]; __syncthreads(); }
    const float rms = rsqrtf(red[0] * (1.f / DMODEL) + EPSV);
#pragma unroll
    for (int j = 0; j < 4; ++j) {
        const int i = t + j * 256;
        out[(size_t)n * DMODEL + i] = (bf16)(v[j] * rms * w[i]);
    }
}

// ---------------------------------------------------------------------------
extern "C" void kernel_launch(void* const* d_in, const int* in_sizes, int n_in,
                              void* d_out, int out_size, void* d_ws, size_t ws_size,
                              hipStream_t stream) {
    (void)in_sizes; (void)n_in; (void)out_size; (void)ws_size;

    const float* s          = (const float*)d_in[0];
    const float* W_in       = (const float*)d_in[1];
    const float* in_norm_w  = (const float*)d_in[2];
    const float* Wr         = (const float*)d_in[3];
    const float* ebias      = (const float*)d_in[4];
    const float* Wg         = (const float*)d_in[5];
    const float* Wu         = (const float*)d_in[6];
    const float* Wd         = (const float*)d_in[7];
    const float* enorm_w    = (const float*)d_in[8];
    const float* out_norm_w = (const float*)d_in[9];
    const float* W_out      = (const float*)d_in[10];
    float* out = (float*)d_out;

    char* w = (char*)d_ws;
    auto carve = [&](size_t bytes) -> char* {
        char* p = w;
        w += (bytes + 255) & ~(size_t)255;
        return p;
    };

    bf16* xb     = (bf16*)carve((size_t)N_TOK * DMODEL * 2);
    bf16* winb   = (bf16*)carve((size_t)DMODEL * DMODEL * 2);
    bf16* wgb    = (bf16*)carve((size_t)NEXP * DFF * DMODEL * 2);
    bf16* wub    = (bf16*)carve((size_t)NEXP * DFF * DMODEL * 2);
    bf16* wdb    = (bf16*)carve((size_t)NEXP * DMODEL * DFF * 2);
    bf16* woutb  = (bf16*)carve((size_t)DMODEL * DMODEL * 2);
    float* hpre  = (float*)carve((size_t)N_TOK * DMODEL * 4);
    bf16* hbf    = (bf16*)carve((size_t)N_TOK * DMODEL * 2);
    int*  counts = (int*)carve(256);
    int*  tok    = (int*)carve((size_t)NEXP * N_TOK * 4);
    float* gate  = (float*)carve((size_t)NEXP * N_TOK * 4);
    bf16* Gbuf   = (bf16*)carve((size_t)N_TOK * DFF * 2);
    float* Ybuf  = (float*)carve((size_t)N_TOK * DMODEL * 4);
    float* outacc= (float*)carve((size_t)N_TOK * DMODEL * 4);
    bf16* zb     = (bf16*)carve((size_t)N_TOK * DMODEL * 2);

    auto cvt = [&](const float* src, bf16* dst, long n) {
        f32_to_bf16_kernel<<<4096, 256, 0, stream>>>(src, dst, n / 4);
    };
    cvt(s,     xb,    (long)N_TOK * DMODEL);
    cvt(W_in,  winb,  (long)DMODEL * DMODEL);
    cvt(Wg,    wgb,   (long)NEXP * DFF * DMODEL);
    cvt(Wu,    wub,   (long)NEXP * DFF * DMODEL);
    cvt(Wd,    wdb,   (long)NEXP * DMODEL * DFF);
    cvt(W_out, woutb, (long)DMODEL * DMODEL);

    zero_f32_kernel<<<2048, 256, 0, stream>>>(outacc, (long)N_TOK * DMODEL);
    zero_f32_kernel<<<1, 32, 0, stream>>>((float*)counts, NEXP);

    // h_pre = x @ W_in^T
    gemm_bf16_kernel<<<dim3(DMODEL / 128, N_TOK / 128), 256, 0, stream>>>(
        xb, DMODEL, winb, DMODEL, hpre, DMODEL, N_TOK, DMODEL, nullptr);

    // h = rmsnorm(h_pre); logits, top-2, gates, per-expert compaction
    rmsnorm_router_kernel<<<N_TOK, 256, 0, stream>>>(
        hpre, in_norm_w, Wr, ebias, hbf, counts, tok, gate);

    for (int e = 0; e < NEXP; ++e) {
        const bf16* wg_e = wgb + (size_t)e * DFF * DMODEL;
        const bf16* wu_e = wub + (size_t)e * DFF * DMODEL;
        const bf16* wd_e = wdb + (size_t)e * DMODEL * DFF;
        moe_gateup_kernel<<<dim3(DFF / 128, N_TOK / 128), 256, 0, stream>>>(
            hbf, wg_e, wu_e, Gbuf, tok + (size_t)e * N_TOK, counts + e);
        gemm_bf16_kernel<<<dim3(DMODEL / 128, N_TOK / 128), 256, 0, stream>>>(
            Gbuf, DFF, wd_e, DFF, Ybuf, DMODEL, N_TOK, DFF, counts + e);
        expert_epilogue_kernel<<<N_TOK, 256, 0, stream>>>(
            Ybuf, tok + (size_t)e * N_TOK, gate + (size_t)e * N_TOK,
            counts + e, enorm_w + (size_t)e * DMODEL, outacc);
    }

    // s_hat = rmsnorm(out) @ W_out^T
    rmsnorm_to_bf16_kernel<<<N_TOK, 256, 0, stream>>>(outacc, out_norm_w, zb);
    gemm_bf16_kernel<<<dim3(DMODEL / 128, N_TOK / 128), 256, 0, stream>>>(
        zb, DMODEL, woutb, DMODEL, out, DMODEL, N_TOK, DMODEL, nullptr);
}
